// StructureBasedNeuralTangentKernel_3083786519332
// MI455X (gfx1250) — matrix operand order repaired and checked by reading, not compiled, not run on hardware
//
#include <hip/hip_runtime.h>
#include <hip/hip_bf16.h>
#include <math.h>

// ---------------------------------------------------------------------------
// StructureBasedNeuralTangentKernel for MI455X (gfx1250).
// 16 GEMMs of 2048^3 + 3 GEMMs of 2048^2*256 via v_wmma_f32_16x16x32_bf16.
// Working set is L2-resident (192MB). GEMM: 64x32 wave tiles (4x2 WMMA accs,
// 1.5 b128-loads per WMMA) with explicit double-buffered fragment loads so a
// full k-slice of loads is in flight while the previous slice runs on the
// matrix pipe.
// ---------------------------------------------------------------------------

#define SB_N 2048
#define SB_D 256
#define SB_PI 3.14159265358979323846f

typedef __attribute__((ext_vector_type(16))) __bf16 v16bf;
typedef __attribute__((ext_vector_type(8)))  __bf16 v8bf;
typedef __attribute__((ext_vector_type(8)))  float  v8f;

// ---- WMMA fragment loaders (wave32) ---------------------------------------
// A fragment: 16x32 (MxK), row-major, stride ld (elements).
// ISA layout: lanes 0-15 = rows M=0..15 holding K 0..7 (v0-3) and 16..23 (v4-7);
// lanes 16-31 = same rows holding K 8..15 and 24..31.
__device__ __forceinline__ v16bf load_a_frag(const __bf16* __restrict__ A, int ld) {
  const int lane = threadIdx.x & 31;
  const int row  = lane & 15;
  const int kh   = (lane >> 4) << 3;            // 0 or 8
  const __bf16* p = A + (size_t)row * ld + kh;
  v8bf lo = *reinterpret_cast<const v8bf*>(p);         // K = kh .. kh+7
  v8bf hi = *reinterpret_cast<const v8bf*>(p + 16);    // K = kh+16 .. kh+23
  return __builtin_shufflevector(lo, hi, 0,1,2,3,4,5,6,7,8,9,10,11,12,13,14,15);
}

// B fragment: 32x16 (KxN) supplied as Bt (NxK row-major, K contiguous, stride ld).
// Layout (per sparse-B pattern scaled to K=32): lane&15 = column N,
// lanes 0-15 hold K 0..7 and 16..23, lanes 16-31 hold K 8..15 and 24..31.
__device__ __forceinline__ v16bf load_b_frag(const __bf16* __restrict__ Bt, int ld) {
  const int lane = threadIdx.x & 31;
  const int col  = lane & 15;
  const int kh   = (lane >> 4) << 3;            // 0 or 8
  const __bf16* p = Bt + (size_t)col * ld + kh;
  v8bf lo = *reinterpret_cast<const v8bf*>(p);
  v8bf hi = *reinterpret_cast<const v8bf*>(p + 16);
  return __builtin_shufflevector(lo, hi, 0,1,2,3,4,5,6,7,8,9,10,11,12,13,14,15);
}

__device__ __forceinline__ void load_frags(const __bf16* __restrict__ Abase,
                                           const __bf16* __restrict__ Bbase,
                                           int K, int kk,
                                           v16bf (&a)[4], v16bf (&b)[2]) {
#pragma unroll
  for (int i = 0; i < 4; ++i)
    a[i] = load_a_frag(Abase + (size_t)(i * 16) * K + kk, K);
#pragma unroll
  for (int j = 0; j < 2; ++j)
    b[j] = load_b_frag(Bbase + (size_t)(j * 16) * K + kk, K);
}

__device__ __forceinline__ void do_wmmas(v8f (&acc)[4][2],
                                         const v16bf (&a)[4], const v16bf (&b)[2]) {
#pragma unroll
  for (int i = 0; i < 4; ++i)
#pragma unroll
    for (int j = 0; j < 2; ++j)
      acc[i][j] = __builtin_amdgcn_wmma_f32_16x16x32_bf16(
          false, a[i], false, b[j], (short)0, acc[i][j], false, false);
}

// ---- NT GEMM: C[2048x2048] (f32) = A (2048xK bf16) * Bt(2048xK bf16)^T ----
// 8 waves/block in a 2x4 grid; each wave owns a 64(M)x32(N) tile
// (4x2 WMMA accumulators). Block tile 128x128. Grid = (16, 16).
// K is a multiple of 64 (256 or 2048).
__global__ void __launch_bounds__(256) k_gemm_nt(
    float* __restrict__ C, const __bf16* __restrict__ A,
    const __bf16* __restrict__ Bt, int K)
{
  const int lane = threadIdx.x & 31;
  const int wid  = threadIdx.x >> 5;   // 0..7
  const int wrow = wid >> 2;           // 0..1
  const int wcol = wid & 3;            // 0..3
  const int m0 = blockIdx.y * 128 + wrow * 64;
  const int n0 = blockIdx.x * 128 + wcol * 32;

  const __bf16* Abase = A  + (size_t)m0 * K;
  const __bf16* Bbase = Bt + (size_t)n0 * K;

  v8f acc[4][2] = {};
  v16bf aA[4], bA[2], aB[4], bB[2];

  // double-buffered k loop, 64 per trip (two 32-wide WMMA slices)
  load_frags(Abase, Bbase, K, 0, aA, bA);
  int kk = 0;
  for (; kk + 64 < K; kk += 64) {
    load_frags(Abase, Bbase, K, kk + 32, aB, bB);
    do_wmmas(acc, aA, bA);
    load_frags(Abase, Bbase, K, kk + 64, aA, bA);
    do_wmmas(acc, aB, bB);
  }
  load_frags(Abase, Bbase, K, kk + 32, aB, bB);
  do_wmmas(acc, aA, bA);
  do_wmmas(acc, aB, bB);

  // C/D layout: VGPR r -> row (lane<16 ? r : 8+r), column = lane&15.
  const int crow = (lane >> 4) * 8;
  const int ccol = lane & 15;
#pragma unroll
  for (int i = 0; i < 4; ++i)
#pragma unroll
    for (int j = 0; j < 2; ++j) {
      float* cp = C + (size_t)(m0 + i * 16 + crow) * SB_N + (n0 + j * 16 + ccol);
#pragma unroll
      for (int r = 0; r < 8; ++r)
        cp[(size_t)r * SB_N] = acc[i][j][r];
    }
}

// ---- f32 -> bf16 convert (grid-stride) ------------------------------------
__global__ void k_cvt_bf16(__bf16* __restrict__ dst, const float* __restrict__ src, int n) {
  for (int i = blockIdx.x * blockDim.x + threadIdx.x; i < n; i += gridDim.x * blockDim.x)
    dst[i] = (__bf16)src[i];
}

// ---- transpose + convert: dst[j*N+i] = bf16(src[i*N+j]) (32x32 LDS tiles) --
__global__ void k_transpose_cvt(__bf16* __restrict__ dst, const float* __restrict__ src) {
  __shared__ float t[32][33];
  const int bx = blockIdx.x, by = blockIdx.y;
  const int tx = threadIdx.x;
#pragma unroll
  for (int r = threadIdx.y; r < 32; r += 8)
    t[r][tx] = src[(size_t)(by * 32 + r) * SB_N + bx * 32 + tx];
  __syncthreads();
#pragma unroll
  for (int r = threadIdx.y; r < 32; r += 8)
    dst[(size_t)(bx * 32 + r) * SB_N + by * 32 + tx] = (__bf16)t[tx][r];
}

// ---- d[i] = sqrt(S[i,i]) ---------------------------------------------------
__global__ void k_diag_sqrt(float* __restrict__ d, const float* __restrict__ S) {
  int i = blockIdx.x * blockDim.x + threadIdx.x;
  if (i < SB_N) d[i] = sqrtf(S[(size_t)i * (SB_N + 1)]);
}

// ---- arccos kernel helpers -------------------------------------------------
__device__ __forceinline__ void arccos_kernel(float sn, float& nw, float& deg) {
  sn = fminf(fmaxf(sn, -0.9999f), 0.9999f);
  float ac = acosf(sn);
  nw  = (sn * (SB_PI - ac) + sqrtf(1.0f - sn * sn)) * (1.0f / SB_PI);
  deg = (SB_PI - ac) * (1.0f / SB_PI);
}

// ---- S = arccos_kernel(S / d_i d_j) * d_i d_j ------------------------------
__global__ void k_update_diag(float* __restrict__ S, const float* __restrict__ d) {
  int idx = blockIdx.x * blockDim.x + threadIdx.x;
  if (idx >= SB_N * SB_N) return;
  int i = idx >> 11, j = idx & (SB_N - 1);
  float dd = d[i] * d[j];
  float nw, deg;
  arccos_kernel(S[idx] / dd, nw, deg);
  S[idx] = nw * dd;
}

// ---- fused: sigma update + theta = theta*deg + sigma -----------------------
__global__ void k_update_sigma_theta(float* __restrict__ sig, float* __restrict__ th,
                                     const float* __restrict__ d1, const float* __restrict__ d2) {
  int idx = blockIdx.x * blockDim.x + threadIdx.x;
  if (idx >= SB_N * SB_N) return;
  int i = idx >> 11, j = idx & (SB_N - 1);
  float dd = d1[i] * d2[j];
  float nw, deg;
  arccos_kernel(sig[idx] / dd, nw, deg);
  float ns = nw * dd;
  sig[idx] = ns;
  th[idx]  = th[idx] * deg + ns;
}

// ---------------------------------------------------------------------------
extern "C" void kernel_launch(void* const* d_in, const int* in_sizes, int n_in,
                              void* d_out, int out_size, void* d_ws, size_t ws_size,
                              hipStream_t stream) {
  (void)in_sizes; (void)n_in; (void)out_size; (void)ws_size;
  const float* g1 = (const float*)d_in[0];
  const float* g2 = (const float*)d_in[1];
  const float* A1 = (const float*)d_in[2];
  const float* A2 = (const float*)d_in[3];
  float* TH = (float*)d_out;

  const size_t NN = (size_t)SB_N * SB_N;
  // workspace layout (~64MB + 32KB)
  __bf16* A1b = (__bf16*)d_ws;
  __bf16* A2b = A1b + NN;
  __bf16* Xb  = A2b + NN;       // first-operand staging (bf16)
  __bf16* Yb  = Xb + NN;        // second-operand staging (bf16)
  float*  S   = (float*)(Yb + NN);  // sigma scratch (diag phases) / SIG (main)
  float*  P   = S + NN;             // aggr intermediate
  float*  dv  = P + NN;             // 4 diag vectors
  float* d1v[2] = {dv, dv + SB_N};
  float* d2v[2] = {dv + 2 * SB_N, dv + 3 * SB_N};

  const dim3 gemm_grid(SB_N / 128, SB_N / 128);
  const dim3 ew_grid((unsigned)((NN + 255) / 256));
  const dim3 tr_grid(SB_N / 32, SB_N / 32), tr_blk(32, 8);

  auto gemm = [&](float* C, const __bf16* A, const __bf16* Bt, int K) {
    k_gemm_nt<<<gemm_grid, 256, 0, stream>>>(C, A, Bt, K);
  };
  auto cvt = [&](__bf16* dst, const float* src, size_t n) {
    k_cvt_bf16<<<dim3((unsigned)((n + 255) / 256)), 256, 0, stream>>>(dst, src, (int)n);
  };
  // one aggr step: OUT(f32) = Ab @ IN @ Bb^T   (IN f32, Ab/Bb bf16 adjacency)
  auto aggr = [&](float* OUT, const float* IN, const __bf16* Ab, const __bf16* Bb) {
    k_transpose_cvt<<<tr_grid, tr_blk, 0, stream>>>(Yb, IN);   // Yb = bf16(IN^T)
    gemm(P, Ab, Yb, SB_N);                                     // P = Ab @ IN
    cvt(Xb, P, NN);
    gemm(OUT, Xb, Bb, SB_N);                                   // OUT = P @ Bb^T
  };

  // ---- adjacency to bf16 (binary: exact) ----
  cvt(A1b, A1, NN);
  cvt(A2b, A2, NN);

  // ---- diag lists ----
  const __bf16* Ab[2] = {A1b, A2b};
  const float*  gg[2] = {g1, g2};
  float* dvk[2][2] = {{d1v[0], d1v[1]}, {d2v[0], d2v[1]}};
  for (int s = 0; s < 2; ++s) {
    cvt(Xb, gg[s], (size_t)SB_N * SB_D);
    gemm(S, Xb, Xb, SB_D);                                     // S = g g^T
    for (int k = 0; k < 2; ++k) {
      aggr(S, S, Ab[s], Ab[s]);
      k_diag_sqrt<<<dim3(SB_N / 256), 256, 0, stream>>>(dvk[s][k], S);
      k_update_diag<<<ew_grid, 256, 0, stream>>>(S, dvk[s][k]);
    }
  }

  // ---- main: sigma = g1 g2^T, theta = sigma ----
  cvt(Xb, g1, (size_t)SB_N * SB_D);
  cvt(Yb, g2, (size_t)SB_N * SB_D);
  gemm(S, Xb, Yb, SB_D);                                       // S = sigma
  hipMemcpyAsync(TH, S, NN * sizeof(float), hipMemcpyDeviceToDevice, stream);

  for (int k = 0; k < 2; ++k) {
    aggr(S, S, A1b, A2b);                                      // sigma aggr
    aggr(TH, TH, A1b, A2b);                                    // theta aggr
    for (int l = 0; l < 2; ++l)                                // L=2 updates
      k_update_sigma_theta<<<ew_grid, 256, 0, stream>>>(S, TH, d1v[k], d2v[k]);
  }
}
